// GPT_69372311765565
// MI455X (gfx1250) — matrix operand order repaired and checked by reading, not compiled, not run on hardware
//
#include <hip/hip_runtime.h>

#define D_MODEL 1024
#define NBLK 6
#define NHEAD 16
#define HDIM 64
#define BATCH 2
#define SEQ 1024
#define ROWS (BATCH * SEQ)   // 2048
#define FF 4096
#define VOCAB 32000
#define LN_EPS 1e-5f

typedef __attribute__((ext_vector_type(16))) __bf16 v16bf;
typedef __attribute__((ext_vector_type(8)))  __bf16 v8bf;
typedef __attribute__((ext_vector_type(8)))  float  v8f;
typedef __attribute__((ext_vector_type(4)))  int    v4i;

#define AS1 __attribute__((address_space(1)))
#define AS3 __attribute__((address_space(3)))

#if __has_builtin(__builtin_amdgcn_global_load_async_to_lds_b128)
#define HAVE_ASYNC_LDS 1
#else
#define HAVE_ASYNC_LDS 0
#endif

// ---------------- bf16 helpers (manual, storage-only __bf16) ----------------
static __device__ __forceinline__ float bf2f(__bf16 h) {
  unsigned short u = __builtin_bit_cast(unsigned short, h);
  unsigned int x = ((unsigned int)u) << 16;
  return __builtin_bit_cast(float, x);
}
static __device__ __forceinline__ __bf16 f2bf(float f) {
  unsigned int x = __builtin_bit_cast(unsigned int, f);
  x += 0x7FFFu + ((x >> 16) & 1u);          // round-to-nearest-even
  unsigned short u = (unsigned short)(x >> 16);
  return __builtin_bit_cast(__bf16, u);
}
static __device__ __forceinline__ float bflo(unsigned int u) {
  return __builtin_bit_cast(float, u << 16);
}
static __device__ __forceinline__ float bfhi(unsigned int u) {
  return __builtin_bit_cast(float, u & 0xFFFF0000u);
}

// async-copy 16B global(bf16) -> LDS, plus wait helpers
static __device__ __forceinline__ void copy16_g2l(const __bf16* src, __bf16* dst) {
#if HAVE_ASYNC_LDS
  __builtin_amdgcn_global_load_async_to_lds_b128(
      (AS1 v4i*)src, (AS3 v4i*)dst, 0, 0);
#else
  *(uint4*)dst = *(const uint4*)src;
#endif
}
static __device__ __forceinline__ void wait_async_le2() {
#if HAVE_ASYNC_LDS
#if __has_builtin(__builtin_amdgcn_s_wait_asynccnt)
  __builtin_amdgcn_s_wait_asynccnt(2);
#else
  asm volatile("s_wait_asynccnt 0x2" ::: "memory");
#endif
#endif
}
static __device__ __forceinline__ void wait_async_le0() {
#if HAVE_ASYNC_LDS
#if __has_builtin(__builtin_amdgcn_s_wait_asynccnt)
  __builtin_amdgcn_s_wait_asynccnt(0);
#else
  asm volatile("s_wait_asynccnt 0x0" ::: "memory");
#endif
#endif
}

// ---------------- embedding: x = tok_emb[ctx] + pos_emb ----------------
__global__ __launch_bounds__(256) void embed_kernel(
    const int* __restrict__ ctx, const float* __restrict__ tok,
    const float* __restrict__ pos, float* __restrict__ x) {
  int row = blockIdx.x;
  int s = row & (SEQ - 1);
  int t = ctx[row];
  const float* tr = tok + (size_t)t * D_MODEL;
  const float* pr = pos + (size_t)s * D_MODEL;
  float* xr = x + (size_t)row * D_MODEL;
  for (int i = threadIdx.x; i < D_MODEL; i += 256)
    xr[i] = tr[i] + pr[i];
}

// ---------------- layernorm (fp32 in -> bf16 out) ----------------
static __device__ __forceinline__ float block_sum256(float v, float* red) {
  #pragma unroll
  for (int m = 16; m >= 1; m >>= 1) v += __shfl_xor(v, m, 32);
  int wid = threadIdx.x >> 5;
  if ((threadIdx.x & 31) == 0) red[wid] = v;
  __syncthreads();
  if (threadIdx.x == 0) {
    float s = 0.f;
    #pragma unroll
    for (int i = 0; i < 8; ++i) s += red[i];
    red[0] = s;
  }
  __syncthreads();
  float r = red[0];
  __syncthreads();
  return r;
}

__global__ __launch_bounds__(256) void ln_kernel(
    const float* __restrict__ x, const float* __restrict__ g,
    const float* __restrict__ b, __bf16* __restrict__ out) {
  __shared__ float red[8];
  int row = blockIdx.x;
  const float* xr = x + (size_t)row * D_MODEL;
  float v[4];
  float s = 0.f;
  #pragma unroll
  for (int i = 0; i < 4; ++i) { v[i] = xr[threadIdx.x + 256 * i]; s += v[i]; }
  s = block_sum256(s, red);
  float mu = s * (1.f / D_MODEL);
  float q = 0.f;
  #pragma unroll
  for (int i = 0; i < 4; ++i) { float d = v[i] - mu; q += d * d; }
  q = block_sum256(q, red);
  float rinv = rsqrtf(q * (1.f / D_MODEL) + LN_EPS);
  __bf16* orow = out + (size_t)row * D_MODEL;
  #pragma unroll
  for (int i = 0; i < 4; ++i) {
    int d = threadIdx.x + 256 * i;
    orow[d] = f2bf((v[i] - mu) * rinv * g[d] + b[d]);
  }
}

// ---------------- WMMA GEMM: C = A(bf16)[M,K] * B(f32->bf16)[K,N] ----------------
// 128x128 block tile, 8 waves, each wave 32(M) x 64(N) = 2x4 wmma tiles, K-step 32.
// Double-buffered LDS; A tile moved with GLOBAL_LOAD_ASYNC_TO_LDS_B128 (ASYNCcnt),
// B tile register-staged one stage ahead (fp32 -> bf16 convert, stored transposed).
#define TM 128
#define TN 128
#define TK 32
#define LSTR 40   // LDS row stride in halfs (32 + 8 pad), rows stay 16B aligned

template <bool HAS_BIAS, bool RELU, bool RESIDUAL, bool OUT_BF16, bool OUT_F32>
__global__ __launch_bounds__(256) void gemm_kernel(
    const __bf16* __restrict__ A, const float* __restrict__ Bw,
    const float* __restrict__ bias, float* __restrict__ Cf32,
    __bf16* __restrict__ Cbf, int M, int N, int K) {
  __shared__ __bf16 As[2][TM * LSTR];
  __shared__ __bf16 Bt[2][TN * LSTR];   // B tile stored transposed: [n][k]

  const int n0 = blockIdx.x * TN;
  const int m0 = blockIdx.y * TM;
  const int t = threadIdx.x;
  const int lane = t & 31;
  const int wave = t >> 5;
  const int wm = (wave & 3) * 32;   // wave M offset in tile
  const int wn = (wave >> 2) * 64;  // wave N offset in tile
  const int lm = lane & 15;
  const int lk = lane >> 4;

  v8f acc[2][4] = {};
  float4 breg[4];

  // per-thread tile-copy geometry
  const int ar = (t + 0) >> 2;             // A: 2 x 16B chunks per thread
  const int ac = (t & 3) << 3;
  const int ar2 = (t + 256) >> 2;
  // B: 4 x float4 per thread
  // e = t + 256*i ; kk = e>>5 ; c = (e&31)*4

  // issue A(0) async copies
  copy16_g2l(A + (size_t)(m0 + ar) * K + 0 + ac, &As[0][ar * LSTR + ac]);
  copy16_g2l(A + (size_t)(m0 + ar2) * K + 0 + ac, &As[0][ar2 * LSTR + ac]);
  // load + convert + store B(0)
  #pragma unroll
  for (int i = 0; i < 4; ++i) {
    int e = t + 256 * i;
    int kk = e >> 5;
    int c = (e & 31) << 2;
    breg[i] = *(const float4*)(Bw + (size_t)kk * N + n0 + c);
  }
  #pragma unroll
  for (int i = 0; i < 4; ++i) {
    int e = t + 256 * i;
    int kk = e >> 5;
    int c = (e & 31) << 2;
    Bt[0][(c + 0) * LSTR + kk] = f2bf(breg[i].x);
    Bt[0][(c + 1) * LSTR + kk] = f2bf(breg[i].y);
    Bt[0][(c + 2) * LSTR + kk] = f2bf(breg[i].z);
    Bt[0][(c + 3) * LSTR + kk] = f2bf(breg[i].w);
  }

  const int nsteps = K / TK;
  for (int s = 0; s < nsteps; ++s) {
    const int cur = s & 1;
    const int nxt = cur ^ 1;
    const bool has_next = (s + 1) < nsteps;

    if (has_next) {
      const int k1 = (s + 1) * TK;
      copy16_g2l(A + (size_t)(m0 + ar) * K + k1 + ac, &As[nxt][ar * LSTR + ac]);
      copy16_g2l(A + (size_t)(m0 + ar2) * K + k1 + ac, &As[nxt][ar2 * LSTR + ac]);
      #pragma unroll
      for (int i = 0; i < 4; ++i) {
        int e = t + 256 * i;
        int kk = e >> 5;
        int c = (e & 31) << 2;
        breg[i] = *(const float4*)(Bw + (size_t)(k1 + kk) * N + n0 + c);
      }
      wait_async_le2();      // the 2 older ops (stage s A tile) have landed
    } else {
      wait_async_le0();
    }
    __syncthreads();

    // A fragments: lane lm = M row, lk selects K-half per ISA 16-bit A layout.
    v16bf af[2];
    #pragma unroll
    for (int mi = 0; mi < 2; ++mi) {
      const __bf16* p = &As[cur][(wm + mi * 16 + lm) * LSTR + lk * 8];
      v8bf lo = *(const v8bf*)(p);        // K = lk*8 + 0..7
      v8bf hi = *(const v8bf*)(p + 16);   // K = 16 + lk*8 + 0..7
      af[mi] = __builtin_shufflevector(lo, hi, 0, 1, 2, 3, 4, 5, 6, 7,
                                       8, 9, 10, 11, 12, 13, 14, 15);
    }
    #pragma unroll
    for (int ni = 0; ni < 4; ++ni) {
      // B fragment: lane lm = N col, K = lk*8 + 0..7 then 16 + lk*8 + 0..7 (matches A)
      const __bf16* p = &Bt[cur][(wn + ni * 16 + lm) * LSTR + lk * 8];
      v8bf lo = *(const v8bf*)(p);
      v8bf hi = *(const v8bf*)(p + 16);
      v16bf bfrag = __builtin_shufflevector(lo, hi, 0, 1, 2, 3, 4, 5, 6, 7,
                                            8, 9, 10, 11, 12, 13, 14, 15);
      #pragma unroll
      for (int mi = 0; mi < 2; ++mi) {
        acc[mi][ni] = __builtin_amdgcn_wmma_f32_16x16x32_bf16(
            false, af[mi], false, bfrag, (short)0, acc[mi][ni], false, false);
      }
    }
    __syncthreads();   // all waves done reading cur buffers

    if (has_next) {    // stores complete before next compute barrier (dscnt wait)
      #pragma unroll
      for (int i = 0; i < 4; ++i) {
        int e = t + 256 * i;
        int kk = e >> 5;
        int c = (e & 31) << 2;
        Bt[nxt][(c + 0) * LSTR + kk] = f2bf(breg[i].x);
        Bt[nxt][(c + 1) * LSTR + kk] = f2bf(breg[i].y);
        Bt[nxt][(c + 2) * LSTR + kk] = f2bf(breg[i].z);
        Bt[nxt][(c + 3) * LSTR + kk] = f2bf(breg[i].w);
      }
    }
  }

  // Epilogue: C/D layout -> row = r + lk*8, col = lm (within each 16x16 tile)
  #pragma unroll
  for (int mi = 0; mi < 2; ++mi) {
    #pragma unroll
    for (int ni = 0; ni < 4; ++ni) {
      #pragma unroll
      for (int r = 0; r < 8; ++r) {
        int row = m0 + wm + mi * 16 + lk * 8 + r;
        int col = n0 + wn + ni * 16 + lm;
        float v = acc[mi][ni][r];
        if (HAS_BIAS) v += bias[col];
        if (RELU) v = v > 0.f ? v : 0.f;
        size_t idx = (size_t)row * N + col;
        if (RESIDUAL) v += Cf32[idx];
        if (OUT_F32) Cf32[idx] = v;
        if (OUT_BF16) Cbf[idx] = f2bf(v);
      }
    }
  }
}

// ---------------- causal attention, flash-style, one wave per query row ----------------
__global__ __launch_bounds__(128) void attn_kernel(
    const __bf16* __restrict__ Q, const __bf16* __restrict__ Km,
    const __bf16* __restrict__ V, __bf16* __restrict__ O) {
  __shared__ float sq[4][HDIM];
  const int lane = threadIdx.x & 31;
  const int wid = threadIdx.x >> 5;
  const int w = blockIdx.x * 4 + wid;       // global wave id
  const int qi = w & (SEQ - 1);
  const int bh = w >> 10;                   // SEQ == 1024
  const int h = bh & (NHEAD - 1);
  const int b = bh >> 4;                    // NHEAD == 16

  const size_t base = (size_t)(b * SEQ) * D_MODEL + (size_t)h * HDIM;
  // stage this wave's q row (fp32) in LDS
  const unsigned int* qp = (const unsigned int*)(Q + base + (size_t)qi * D_MODEL);
  unsigned int qu = qp[lane];
  sq[wid][2 * lane] = bflo(qu);
  sq[wid][2 * lane + 1] = bfhi(qu);
  __syncthreads();

  float m = -3.0e38f, lsum = 0.f, o0 = 0.f, o1 = 0.f;
  const float scale = 0.125f;               // 1/sqrt(64)

  for (int j0 = 0; j0 <= qi; j0 += 32) {
    int j = j0 + lane;                      // each lane owns one key
    float s = -3.0e38f;
    if (j <= qi) {
      const unsigned int* kp = (const unsigned int*)(Km + base + (size_t)j * D_MODEL);
      float acc = 0.f;
      #pragma unroll
      for (int dd = 0; dd < 32; ++dd) {
        unsigned int ku = kp[dd];
        acc += sq[wid][2 * dd] * bflo(ku) + sq[wid][2 * dd + 1] * bfhi(ku);
      }
      s = acc * scale;
    }
    float cm = s;
    #pragma unroll
    for (int msk = 16; msk >= 1; msk >>= 1) cm = fmaxf(cm, __shfl_xor(cm, msk, 32));
    float mnew = fmaxf(m, cm);
    float corr = __expf(m - mnew);          // 0 on first chunk
    float p = (j <= qi) ? __expf(s - mnew) : 0.f;
    float ps = p;
    #pragma unroll
    for (int msk = 16; msk >= 1; msk >>= 1) ps += __shfl_xor(ps, msk, 32);
    lsum = lsum * corr + ps;
    o0 *= corr;
    o1 *= corr;
    int nk = qi - j0 + 1;
    if (nk > 32) nk = 32;
    for (int tt = 0; tt < nk; ++tt) {
      float pt = __shfl(p, tt, 32);
      const unsigned int* vp =
          (const unsigned int*)(V + base + (size_t)(j0 + tt) * D_MODEL);
      unsigned int vu = vp[lane];           // lane covers dims 2*lane, 2*lane+1
      o0 += pt * bflo(vu);
      o1 += pt * bfhi(vu);
    }
    m = mnew;
  }
  float inv = 1.f / lsum;
  __bf16* orow = O + base + (size_t)qi * D_MODEL;
  orow[2 * lane] = f2bf(o0 * inv);
  orow[2 * lane + 1] = f2bf(o1 * inv);
}

// ---------------- orchestration ----------------
extern "C" void kernel_launch(void* const* d_in, const int* in_sizes, int n_in,
                              void* d_out, int out_size, void* d_ws, size_t ws_size,
                              hipStream_t stream) {
  (void)in_sizes; (void)n_in; (void)out_size; (void)ws_size;

  const int*   ctx     = (const int*)d_in[0];
  const float* tok_emb = (const float*)d_in[1];
  const float* pos_emb = (const float*)d_in[2];
  const float* Wq      = (const float*)d_in[3];
  const float* Wk      = (const float*)d_in[4];
  const float* Wv      = (const float*)d_in[5];
  const float* Wo      = (const float*)d_in[6];
  const float* bo      = (const float*)d_in[7];
  const float* ln1_g   = (const float*)d_in[8];
  const float* ln1_b   = (const float*)d_in[9];
  const float* ln2_g   = (const float*)d_in[10];
  const float* ln2_b   = (const float*)d_in[11];
  const float* W1      = (const float*)d_in[12];
  const float* b1      = (const float*)d_in[13];
  const float* W2      = (const float*)d_in[14];
  const float* b2      = (const float*)d_in[15];
  const float* lnf_g   = (const float*)d_in[16];
  const float* lnf_b   = (const float*)d_in[17];
  const float* Wp      = (const float*)d_in[18];
  const float* bp      = (const float*)d_in[19];

  char* ws = (char*)d_ws;
  float*  x  = (float*)ws;  ws += (size_t)ROWS * D_MODEL * sizeof(float);   // residual stream
  __bf16* hb = (__bf16*)ws; ws += (size_t)ROWS * D_MODEL * 2;               // LN output
  __bf16* qb = (__bf16*)ws; ws += (size_t)ROWS * D_MODEL * 2;
  __bf16* kb = (__bf16*)ws; ws += (size_t)ROWS * D_MODEL * 2;
  __bf16* vb = (__bf16*)ws; ws += (size_t)ROWS * D_MODEL * 2;
  __bf16* ab = (__bf16*)ws; ws += (size_t)ROWS * D_MODEL * 2;               // attention out
  __bf16* fb = (__bf16*)ws; ws += (size_t)ROWS * FF * 2;                    // FFN hidden

  const dim3 gD(D_MODEL / TN, ROWS / TM);   // (8, 16)
  const dim3 gF(FF / TN, ROWS / TM);        // (32, 16)
  const dim3 gV(VOCAB / TN, ROWS / TM);     // (250, 16)

  embed_kernel<<<ROWS, 256, 0, stream>>>(ctx, tok_emb, pos_emb, x);

  for (int L = 0; L < NBLK; ++L) {
    const float* wq = Wq + (size_t)L * D_MODEL * D_MODEL;
    const float* wk = Wk + (size_t)L * D_MODEL * D_MODEL;
    const float* wv = Wv + (size_t)L * D_MODEL * D_MODEL;
    const float* wo = Wo + (size_t)L * D_MODEL * D_MODEL;
    const float* w1 = W1 + (size_t)L * D_MODEL * FF;
    const float* w2 = W2 + (size_t)L * FF * D_MODEL;

    ln_kernel<<<ROWS, 256, 0, stream>>>(x, ln1_g + L * D_MODEL, ln1_b + L * D_MODEL, hb);
    gemm_kernel<false, false, false, true, false><<<gD, 256, 0, stream>>>(
        hb, wq, nullptr, nullptr, qb, ROWS, D_MODEL, D_MODEL);
    gemm_kernel<false, false, false, true, false><<<gD, 256, 0, stream>>>(
        hb, wk, nullptr, nullptr, kb, ROWS, D_MODEL, D_MODEL);
    gemm_kernel<false, false, false, true, false><<<gD, 256, 0, stream>>>(
        hb, wv, nullptr, nullptr, vb, ROWS, D_MODEL, D_MODEL);
    attn_kernel<<<(BATCH * NHEAD * SEQ) / 4, 128, 0, stream>>>(qb, kb, vb, ab);
    gemm_kernel<true, false, true, false, true><<<gD, 256, 0, stream>>>(
        ab, wo, bo + (size_t)L * D_MODEL, x, nullptr, ROWS, D_MODEL, D_MODEL);

    ln_kernel<<<ROWS, 256, 0, stream>>>(x, ln2_g + L * D_MODEL, ln2_b + L * D_MODEL, hb);
    gemm_kernel<true, true, false, true, false><<<gF, 256, 0, stream>>>(
        hb, w1, b1 + (size_t)L * FF, nullptr, fb, ROWS, FF, D_MODEL);
    gemm_kernel<true, false, true, false, true><<<gD, 256, 0, stream>>>(
        fb, w2, b2 + (size_t)L * D_MODEL, x, nullptr, ROWS, D_MODEL, FF);
  }

  ln_kernel<<<ROWS, 256, 0, stream>>>(x, lnf_g, lnf_b, hb);
  gemm_kernel<true, false, false, false, true><<<gV, 256, 0, stream>>>(
      hb, Wp, bp, (float*)d_out, nullptr, ROWS, VOCAB, D_MODEL);
}
